// NDREncoder_42090679501459
// MI455X (gfx1250) — matrix operand order repaired and checked by reading, not compiled, not run on hardware
//
#include <hip/hip_runtime.h>
#include <hip/hip_bf16.h>
#include <math.h>
#include <stdint.h>

// ---------------- model constants ----------------
#define Bx   128
#define Sx   100
#define Dx   512
#define Hx   8
#define DHx  64
#define DFFx 2048
#define NL   6
#define RL   (2*Sx - 1)     // 199
#define PLD  208            // padded ld for pos matrix
#define BS   (Bx*Sx)        // 12800
#define ATT_SCALE 0.125f    // 1/sqrt(64)
#define LOG10000 9.2103403719761836f

// GEMM tiling
#define BM 128
#define BN 64
#define BK 32
#define APITCH (BK + 8)     // 40 elems -> 80B row pitch, 16B aligned

typedef __bf16 bf16_t;
typedef __bf16 v16bf __attribute__((ext_vector_type(16)));
typedef __bf16 v8bf  __attribute__((ext_vector_type(8)));
typedef float  v8f   __attribute__((ext_vector_type(8)));

// async global->LDS copy of 16 bytes (VGLOBAL GLOBAL_LOAD_ASYNC_TO_LDS_B128)
#define ASYNC_CP16(ldsoff, gptr)                                              \
    asm volatile("global_load_async_to_lds_b128 %0, %1, off"                  \
                 :: "v"(ldsoff), "v"(gptr) : "memory")

__device__ __forceinline__ int imin(int a, int b) { return a < b ? a : b; }

// =====================================================================
// Batched/strided bf16 GEMM, f32 accumulate, V_WMMA_F32_16X16X32_BF16.
//   C[z] = A[z] * op(B[z]),  z = b*Hb + h.
// Block tile 128x64, BK=32, 256 threads = 8 waves; wave owns 32x32 of C
// (4 wmma per K-step). Double-buffered LDS, one barrier per K-step.
// Interior tiles (always transB after weight pre-transposition) stage
// A and swizzled-B via GLOBAL_LOAD_ASYNC_TO_LDS_B128 (ASYNCcnt).
// Guarded path is branchless: clamped addresses + cndmask zero-select.
// =====================================================================
__global__ __launch_bounds__(256) void gemm_bf16(
    const bf16_t* __restrict__ A, const bf16_t* __restrict__ B,
    float* __restrict__ C,
    int M, int N, int K, int lda, int ldb, int ldc,
    long long sAb, long long sAh, long long sBb, long long sBh,
    long long sCb, long long sCh, int Hb, int transB)
{
    __shared__ bf16_t As[2][BM][APITCH];       // 2 x 128 x 32 (+pad)
    __shared__ bf16_t Bs[2][(BN/16) * 32 * 16];// 2 x (4 subtiles x 32 x 16)

    const int z  = blockIdx.z;
    const int bb = z / Hb;
    const int hh = z - bb * Hb;
    const bf16_t* Ab = A + (size_t)bb * sAb + (size_t)hh * sAh;
    const bf16_t* Bb = B + (size_t)bb * sBb + (size_t)hh * sBh;
    float*        Cb = C + (size_t)bb * sCb + (size_t)hh * sCh;

    const int tileM = blockIdx.y * BM;
    const int tileN = blockIdx.x * BN;
    const int tid   = threadIdx.x;
    const int lane  = tid & 31;
    const int wid   = tid >> 5;
    const int wm0   = (wid >> 1) * 32;   // 0,32,64,96
    const int wn0   = (wid & 1) * 32;    // 0,32
    const int s0    = wn0 >> 4;          // first of two 16-col subtiles

    // interior fast path: full tile in-bounds, K multiple of BK, transB
    // (after weight pre-transposition every big GEMM is transB=1)
    const bool interior = (tileM + BM <= M) && (tileN + BN <= N) &&
                          ((K & (BK - 1)) == 0) && (transB != 0);

    v8f acc00 = {0.f,0.f,0.f,0.f,0.f,0.f,0.f,0.f};
    v8f acc01 = acc00, acc10 = acc00, acc11 = acc00;

    // A staging: 128 rows, 2 threads/row, 16 contiguous k-elems per thread
    const int ar = tid >> 1;
    const int ac = (tid & 1) << 4;
    // B staging (k-major): 64 cols, 4 threads/col, 8 contiguous k-elems
    const int bn   = tid & 63;
    const int bkc  = (tid >> 6) << 3;          // 0,8,16,24
    const int bdstOff = (((bn >> 4) * 32 + (bkc >> 4) * 16 + (bn & 15)) << 4)
                        + (bkc & 15);

    const int nsteps = (K + BK - 1) / BK;

    v8bf rA0, rA1, rB;   // guarded-path staging registers

    // branchless guarded loads: clamped addresses, zero-select on OOB
    auto guardedLoad = [&](int kk, v8bf& a0, v8bf& a1, v8bf& bv) {
        const int gm  = tileM + ar;
        const bool rOk = gm < M;
        const size_t rowBase = (size_t)imin(gm, M - 1) * lda;
#pragma unroll
        for (int e = 0; e < 8; ++e) {
            int gk = kk + ac + e;
            bf16_t t = Ab[rowBase + imin(gk, K - 1)];
            a0[e] = (rOk && gk < K) ? t : (bf16_t)0.0f;
        }
#pragma unroll
        for (int e = 0; e < 8; ++e) {
            int gk = kk + ac + 8 + e;
            bf16_t t = Ab[rowBase + imin(gk, K - 1)];
            a1[e] = (rOk && gk < K) ? t : (bf16_t)0.0f;
        }
        const int gn  = tileN + bn;
        const bool nOk = gn < N;
        const int gnc = imin(gn, N - 1);
#pragma unroll
        for (int e = 0; e < 8; ++e) {
            int gk  = kk + bkc + e;
            int gkc = imin(gk, K - 1);
            size_t off = transB ? (size_t)gnc * ldb + gkc
                                : (size_t)gkc * ldb + gnc;
            bf16_t t = Bb[off];
            bv[e] = (nOk && gk < K) ? t : (bf16_t)0.0f;
        }
    };

    // ---- prologue: stage step 0 into buffer 0 ----
    if (interior) {
        unsigned la = (unsigned)(uintptr_t)&As[0][ar][ac];
        const bf16_t* ga = Ab + (size_t)(tileM + ar) * lda + ac;
        ASYNC_CP16(la, (const void*)ga);
        ASYNC_CP16(la + 16u, (const void*)(ga + 8));
        unsigned lb = (unsigned)(uintptr_t)&Bs[0][bdstOff];
        const bf16_t* gb = Bb + (size_t)(tileN + bn) * ldb + bkc;
        ASYNC_CP16(lb, (const void*)gb);
    } else {
        guardedLoad(0, rA0, rA1, rB);
        *(v8bf*)&As[0][ar][ac]     = rA0;
        *(v8bf*)&As[0][ar][ac + 8] = rA1;
        *(v8bf*)&Bs[0][bdstOff]    = rB;
    }

    for (int s = 0; s < nsteps; ++s) {
        const int cur = s & 1;
        const int nxt = cur ^ 1;
        if (interior)
            asm volatile("s_wait_asynccnt 0x0" ::: "memory");
        __syncthreads();

        const bool more = (s + 1) < nsteps;
        const int k1 = (s + 1) * BK;
        if (more) {
            if (interior) {
                unsigned la = (unsigned)(uintptr_t)&As[nxt][ar][ac];
                const bf16_t* ga = Ab + (size_t)(tileM + ar) * lda + (k1 + ac);
                ASYNC_CP16(la, (const void*)ga);
                ASYNC_CP16(la + 16u, (const void*)(ga + 8));
                unsigned lb = (unsigned)(uintptr_t)&Bs[nxt][bdstOff];
                const bf16_t* gb = Bb + (size_t)(tileN + bn) * ldb + (k1 + bkc);
                ASYNC_CP16(lb, (const void*)gb);
            } else {
                guardedLoad(k1, rA0, rA1, rB);
            }
        }

        // ---- fragments from buffer `cur` ----
        const int am  = lane & 15;
        const int akb = (lane >> 4) << 4;
        v8bf lo, hi;
        lo = *(const v8bf*)&As[cur][wm0 + am][akb];
        hi = *(const v8bf*)&As[cur][wm0 + am][akb + 8];
        v16bf af0 = __builtin_shufflevector(lo, hi, 0,1,2,3,4,5,6,7,8,9,10,11,12,13,14,15);
        lo = *(const v8bf*)&As[cur][wm0 + 16 + am][akb];
        hi = *(const v8bf*)&As[cur][wm0 + 16 + am][akb + 8];
        v16bf af1 = __builtin_shufflevector(lo, hi, 0,1,2,3,4,5,6,7,8,9,10,11,12,13,14,15);

        const bf16_t* bsrc0 = &Bs[cur][(s0 * 32 + lane) << 4];
        const bf16_t* bsrc1 = &Bs[cur][((s0 + 1) * 32 + lane) << 4];
        lo = *(const v8bf*)bsrc0; hi = *(const v8bf*)(bsrc0 + 8);
        v16bf bf0 = __builtin_shufflevector(lo, hi, 0,1,2,3,4,5,6,7,8,9,10,11,12,13,14,15);
        lo = *(const v8bf*)bsrc1; hi = *(const v8bf*)(bsrc1 + 8);
        v16bf bf1 = __builtin_shufflevector(lo, hi, 0,1,2,3,4,5,6,7,8,9,10,11,12,13,14,15);

        acc00 = __builtin_amdgcn_wmma_f32_16x16x32_bf16(false, af0, false, bf0, (short)0, acc00, false, false);
        acc01 = __builtin_amdgcn_wmma_f32_16x16x32_bf16(false, af0, false, bf1, (short)0, acc01, false, false);
        acc10 = __builtin_amdgcn_wmma_f32_16x16x32_bf16(false, af1, false, bf0, (short)0, acc10, false, false);
        acc11 = __builtin_amdgcn_wmma_f32_16x16x32_bf16(false, af1, false, bf1, (short)0, acc11, false, false);

        if (more && !interior) {
            *(v8bf*)&As[nxt][ar][ac]     = rA0;
            *(v8bf*)&As[nxt][ar][ac + 8] = rA1;
            *(v8bf*)&Bs[nxt][bdstOff]    = rB;
        }
    }

    // ---- epilogue: C layout VGPR r -> M = r + 8*(lane>>4), N = lane&15 ----
    const int cm0 = tileM + wm0 + ((lane >> 4) << 3);
    const int cn  = tileN + wn0 + (lane & 15);
    if (tileM + BM <= M && tileN + BN <= N) {
#pragma unroll
        for (int r = 0; r < 8; ++r) {
            Cb[(size_t)(cm0 + r) * ldc + cn]           = acc00[r];
            Cb[(size_t)(cm0 + r) * ldc + cn + 16]      = acc01[r];
            Cb[(size_t)(cm0 + 16 + r) * ldc + cn]      = acc10[r];
            Cb[(size_t)(cm0 + 16 + r) * ldc + cn + 16] = acc11[r];
        }
    } else {
#pragma unroll
        for (int r = 0; r < 8; ++r) {
            int gm = cm0 + r;
            if (gm < M) {
                if (cn < N)      Cb[(size_t)gm * ldc + cn]      = acc00[r];
                if (cn + 16 < N) Cb[(size_t)gm * ldc + cn + 16] = acc01[r];
            }
            if (gm + 16 < M) {
                if (cn < N)      Cb[(size_t)(gm + 16) * ldc + cn]      = acc10[r];
                if (cn + 16 < N) Cb[(size_t)(gm + 16) * ldc + cn + 16] = acc11[r];
            }
        }
    }
}

// ---------------- elementwise / small kernels ----------------
__global__ void k_convert(const float* __restrict__ x, bf16_t* __restrict__ y, int n) {
    int i = blockIdx.x * blockDim.x + threadIdx.x;
    if (i < n) y[i] = (bf16_t)x[i];
}

// fp32 (rows x cols) -> bf16 transposed (cols x rows)
__global__ void k_convT(const float* __restrict__ x, bf16_t* __restrict__ y,
                        int rows, int cols) {
    int i = blockIdx.x * blockDim.x + threadIdx.x;
    if (i >= rows * cols) return;
    int r = i / cols, c = i - r * cols;
    y[(size_t)c * rows + r] = (bf16_t)x[i];
}

__global__ void k_embed(const int* __restrict__ ids, const float* __restrict__ emb,
                        float* __restrict__ X) {
    int i = blockIdx.x * blockDim.x + threadIdx.x;
    if (i >= BS * Dx) return;
    int c = i & (Dx - 1);
    int t = i >> 9;
    int s = t % Sx;
    float div = __expf(-(float)(c & ~1) * (LOG10000 / (float)Dx));
    float ang = (float)s * div;
    float pe  = (c & 1) ? __cosf(ang) : __sinf(ang);
    X[i] = emb[(size_t)ids[t] * Dx + c] + pe;
}

__global__ void k_relsin(float* __restrict__ out) {
    int i = blockIdx.x * blockDim.x + threadIdx.x;
    if (i >= RL * Dx) return;
    int c = i & (Dx - 1);
    int r = i >> 9;
    float pos = (float)(Sx - 1 - r);            // arange(S-1, -S, -1)
    float div = __expf(-(float)(c & ~1) * (LOG10000 / (float)Dx));
    float ang = pos * div;
    out[i] = (c & 1) ? __cosf(ang) : __sinf(ang);
}

__global__ void k_quqv(const float* __restrict__ Q, const float* __restrict__ u,
                       const float* __restrict__ v, bf16_t* __restrict__ quh,
                       bf16_t* __restrict__ qvh) {
    int i = blockIdx.x * blockDim.x + threadIdx.x;
    if (i >= BS * Dx) return;
    int hd  = i & (Dx - 1);       // h*DH + d
    float q = Q[i];
    quh[i] = (bf16_t)(q + u[hd]);
    qvh[i] = (bf16_t)(q + v[hd]);
}

__global__ void k_bias_relu_bf16(const float* __restrict__ x, const float* __restrict__ b,
                                 bf16_t* __restrict__ y, int n, int cols) {
    int i = blockIdx.x * blockDim.x + threadIdx.x;
    if (i >= n) return;
    float t = x[i] + b[i % cols];
    y[i] = (bf16_t)fmaxf(t, 0.0f);
}

__global__ void k_bias_tanh(const float* __restrict__ x, const float* __restrict__ b,
                            float* __restrict__ y, int n, int cols) {
    int i = blockIdx.x * blockDim.x + threadIdx.x;
    if (i >= n) return;
    y[i] = tanhf(x[i] + b[i % cols]);
}

__global__ void k_bias_sigm(const float* __restrict__ x, const float* __restrict__ b,
                            float* __restrict__ y, int n, int cols) {
    int i = blockIdx.x * blockDim.x + threadIdx.x;
    if (i >= n) return;
    float t = x[i] + b[i % cols];
    y[i] = 1.0f / (1.0f + __expf(-t));
}

__global__ __launch_bounds__(256) void k_res_ln(const float* __restrict__ x,
                                                const float* __restrict__ a,
                                                const float* __restrict__ g,
                                                const float* __restrict__ bb,
                                                float* __restrict__ net,
                                                bf16_t* __restrict__ neth) {
    int row = blockIdx.x, tid = threadIdx.x;
    size_t base = (size_t)row * Dx;
    float v0 = x[base + tid]       + a[base + tid];
    float v1 = x[base + tid + 256] + a[base + tid + 256];
    __shared__ float s1[256], s2[256];
    s1[tid] = v0 + v1;
    s2[tid] = v0 * v0 + v1 * v1;
    __syncthreads();
    for (int o = 128; o > 0; o >>= 1) {
        if (tid < o) { s1[tid] += s1[tid + o]; s2[tid] += s2[tid + o]; }
        __syncthreads();
    }
    float m    = s1[0] * (1.0f / Dx);
    float var  = s2[0] * (1.0f / Dx) - m * m;
    float rstd = rsqrtf(var + 1e-5f);
    float y0 = (v0 - m) * rstd * g[tid]       + bb[tid];
    float y1 = (v1 - m) * rstd * g[tid + 256] + bb[tid + 256];
    net[base + tid]        = y0;
    net[base + tid + 256]  = y1;
    neth[base + tid]       = (bf16_t)y0;
    neth[base + tid + 256] = (bf16_t)y1;
}

// fused: rel-position gather + scale + key mask + softmax, emit bf16 probs
__global__ __launch_bounds__(128) void k_softmax(const float* __restrict__ content,
                                                 const float* __restrict__ posf,
                                                 const int* __restrict__ src_len,
                                                 bf16_t* __restrict__ att) {
    int idx = blockIdx.x;            // (b*H + h)*S + q
    int q   = idx % Sx;
    int bh  = idx / Sx;
    int b   = bh / Hx;
    int h   = bh - b * Hx;
    int tid = threadIdx.x;
    bool act = tid < Sx;
    float s = -3.0e38f;
    if (act) {
        float c = content[((size_t)bh * Sx + q) * Sx + tid];
        float p = posf[(size_t)h * ((size_t)BS * PLD)
                       + (size_t)(b * Sx + q) * PLD + (Sx - 1 - q + tid)];
        s = (c + p) * ATT_SCALE;
        if (tid >= src_len[b]) s = -1.0e9f;
    }
    __shared__ float red[128];
    red[tid] = s; __syncthreads();
    for (int o = 64; o > 0; o >>= 1) {
        if (tid < o) red[tid] = fmaxf(red[tid], red[tid + o]);
        __syncthreads();
    }
    float mx = red[0]; __syncthreads();
    float e = act ? __expf(s - mx) : 0.0f;
    red[tid] = e; __syncthreads();
    for (int o = 64; o > 0; o >>= 1) {
        if (tid < o) red[tid] += red[tid + o];
        __syncthreads();
    }
    float inv = 1.0f / red[0];
    if (act) att[((size_t)bh * Sx + q) * Sx + tid] = (bf16_t)(e * inv);
}

__global__ void k_gate(float* __restrict__ X, const float* __restrict__ proj,
                       const float* __restrict__ bg, int n) {
    int i = blockIdx.x * blockDim.x + threadIdx.x;
    if (i >= n) return;
    float gv = bg[i];
    X[i] = X[i] * (1.0f - gv) + proj[i] * gv;
}

__global__ void k_copy(const float* __restrict__ x, float* __restrict__ y, int n) {
    int i = blockIdx.x * blockDim.x + threadIdx.x;
    if (i < n) y[i] = x[i];
}

// =====================================================================
extern "C" void kernel_launch(void* const* d_in, const int* in_sizes, int n_in,
                              void* d_out, int out_size, void* d_ws, size_t ws_size,
                              hipStream_t stream)
{
    (void)in_sizes; (void)n_in; (void)out_size; (void)ws_size;
    const int*   ids  = (const int*)  d_in[0];
    const int*   slen = (const int*)  d_in[1];
    const float* emb  = (const float*)d_in[2];
    const float* Wq   = (const float*)d_in[3];
    const float* Wkv  = (const float*)d_in[4];
    const float* Wpos = (const float*)d_in[5];
    const float* Wout = (const float*)d_in[6];
    const float* u    = (const float*)d_in[7];
    const float* v    = (const float*)d_in[8];
    const float* ng   = (const float*)d_in[9];
    const float* nb   = (const float*)d_in[10];
    const float* p1w  = (const float*)d_in[11];
    const float* p1b  = (const float*)d_in[12];
    const float* p2w  = (const float*)d_in[13];
    const float* p2b  = (const float*)d_in[14];
    const float* g1w  = (const float*)d_in[15];
    const float* g1b  = (const float*)d_in[16];
    const float* g2w  = (const float*)d_in[17];
    const float* g2b  = (const float*)d_in[18];

    // -------- workspace bump allocator --------
    char* p = (char*)d_ws;
    auto alloc = [&](size_t bytes) -> char* {
        char* r = p; p += (bytes + 255) & ~(size_t)255; return r;
    };
    float*  Xf      = (float*) alloc((size_t)BS * Dx * 4);
    bf16_t* Xh      = (bf16_t*)alloc((size_t)BS * Dx * 2);
    bf16_t* WqT     = (bf16_t*)alloc((size_t)Dx * Dx * 2);       // (N,K)
    bf16_t* WkvT    = (bf16_t*)alloc((size_t)Dx * 2 * Dx * 2);
    bf16_t* WposT   = (bf16_t*)alloc((size_t)Dx * Dx * 2);
    bf16_t* WoutT   = (bf16_t*)alloc((size_t)Dx * Dx * 2);
    bf16_t* p1T     = (bf16_t*)alloc((size_t)Dx * DFFx * 2);
    bf16_t* p2T     = (bf16_t*)alloc((size_t)DFFx * Dx * 2);
    bf16_t* g1T     = (bf16_t*)alloc((size_t)Dx * Dx * 2);
    bf16_t* g2T     = (bf16_t*)alloc((size_t)Dx * Dx * 2);
    float*  relsinf = (float*) alloc((size_t)RL * Dx * 4);
    bf16_t* relsinh = (bf16_t*)alloc((size_t)RL * Dx * 2);
    float*  kposf   = (float*) alloc((size_t)RL * Dx * 4);
    bf16_t* kposh   = (bf16_t*)alloc((size_t)RL * Dx * 2);
    float*  Qf      = (float*) alloc((size_t)BS * Dx * 4);
    float*  KVf     = (float*) alloc((size_t)BS * 2 * Dx * 4);
    bf16_t* KVh     = (bf16_t*)alloc((size_t)BS * 2 * Dx * 2);
    bf16_t* quh     = (bf16_t*)alloc((size_t)BS * Dx * 2);
    bf16_t* qvh     = (bf16_t*)alloc((size_t)BS * Dx * 2);
    float*  contf   = (float*) alloc((size_t)Bx * Hx * Sx * Sx * 4);
    float*  posfull = (float*) alloc((size_t)Hx * BS * PLD * 4);
    bf16_t* atth    = (bf16_t*)alloc((size_t)Bx * Hx * Sx * Sx * 2);
    float*  attoutf = (float*) alloc((size_t)BS * Dx * 4);
    bf16_t* attouth = (bf16_t*)alloc((size_t)BS * Dx * 2);
    float*  aoutf   = (float*) alloc((size_t)BS * Dx * 4);
    float*  netf    = (float*) alloc((size_t)BS * Dx * 4);
    bf16_t* neth    = (bf16_t*)alloc((size_t)BS * Dx * 2);
    float*  ff1f    = (float*) alloc((size_t)BS * DFFx * 4);
    bf16_t* ff1h    = (bf16_t*)alloc((size_t)BS * DFFx * 2);
    float*  t1f     = (float*) alloc((size_t)BS * Dx * 4);
    float*  t2f     = (float*) alloc((size_t)BS * Dx * 4);
    float*  projf   = (float*) alloc((size_t)BS * Dx * 4);
    bf16_t* gateh   = (bf16_t*)alloc((size_t)BS * Dx * 2);
    float*  bgatef  = (float*) alloc((size_t)BS * Dx * 4);

    auto grid1 = [](int n) { return dim3((unsigned)((n + 255) / 256)); };
    auto gemm = [&](const bf16_t* A, const bf16_t* B, float* C,
                    int M, int N, int K, int lda, int ldb, int ldc,
                    long long sAb, long long sAh, long long sBb, long long sBh,
                    long long sCb, long long sCh, int batch, int Hb, int transB) {
        dim3 g((unsigned)((N + BN - 1) / BN), (unsigned)((M + BM - 1) / BM), (unsigned)batch);
        gemm_bf16<<<g, 256, 0, stream>>>(A, B, C, M, N, K, lda, ldb, ldc,
                                         sAb, sAh, sBb, sBh, sCb, sCh, Hb, transB);
    };

    // -------- weight conversion + transposition (B becomes (N,K)) --------
    k_convT<<<grid1(Dx*Dx),    256, 0, stream>>>(Wq,   WqT,   Dx, Dx);
    k_convT<<<grid1(Dx*2*Dx),  256, 0, stream>>>(Wkv,  WkvT,  Dx, 2*Dx);
    k_convT<<<grid1(Dx*Dx),    256, 0, stream>>>(Wpos, WposT, Dx, Dx);
    k_convT<<<grid1(Dx*Dx),    256, 0, stream>>>(Wout, WoutT, Dx, Dx);
    k_convT<<<grid1(Dx*DFFx),  256, 0, stream>>>(p1w,  p1T,   Dx, DFFx);
    k_convT<<<grid1(DFFx*Dx),  256, 0, stream>>>(p2w,  p2T,   DFFx, Dx);
    k_convT<<<grid1(Dx*Dx),    256, 0, stream>>>(g1w,  g1T,   Dx, Dx);
    k_convT<<<grid1(Dx*Dx),    256, 0, stream>>>(g2w,  g2T,   Dx, Dx);

    // -------- k_pos = sinusoid(rel) @ Wpos --------
    k_relsin<<<grid1(RL*Dx), 256, 0, stream>>>(relsinf);
    k_convert<<<grid1(RL*Dx), 256, 0, stream>>>(relsinf, relsinh, RL*Dx);
    gemm(relsinh, WposT, kposf, RL, Dx, Dx, Dx, Dx, Dx, 0,0,0,0,0,0, 1, 1, 1);
    k_convert<<<grid1(RL*Dx), 256, 0, stream>>>(kposf, kposh, RL*Dx);

    // -------- embedding + positional --------
    k_embed<<<grid1(BS*Dx), 256, 0, stream>>>(ids, emb, Xf);

    for (int layer = 0; layer < NL; ++layer) {
        k_convert<<<grid1(BS*Dx), 256, 0, stream>>>(Xf, Xh, BS*Dx);

        // Q = X@Wq ; KV = X@Wkv  (transposed weights -> transB)
        gemm(Xh, WqT,  Qf,  BS, Dx,   Dx, Dx, Dx, Dx,   0,0,0,0,0,0, 1, 1, 1);
        gemm(Xh, WkvT, KVf, BS, 2*Dx, Dx, Dx, Dx, 2*Dx, 0,0,0,0,0,0, 1, 1, 1);
        k_convert<<<grid1(BS*2*Dx), 256, 0, stream>>>(KVf, KVh, BS*2*Dx);
        k_quqv<<<grid1(BS*Dx), 256, 0, stream>>>(Qf, u, v, quh, qvh);

        // content[b,h,q,k] = (q+u) . k   (batched over b*H, B^T)
        gemm(quh, KVh, contf, Sx, Sx, DHx,
             Dx, 2*Dx, Sx,
             (long long)Sx*Dx, DHx,
             (long long)Sx*2*Dx, 2*DHx,
             (long long)Hx*Sx*Sx, (long long)Sx*Sx,
             Bx*Hx, Hx, 1);

        // posfull[h, b*S+q, r] = (q+v) . k_pos[r]   (batched over H, B^T)
        gemm(qvh, kposh, posfull, BS, RL, DHx,
             Dx, Dx, PLD,
             0, DHx, 0, DHx, 0, (long long)BS*PLD,
             Hx, Hx, 1);

        // gather + scale + mask + softmax -> bf16 probs
        k_softmax<<<dim3(Bx*Hx*Sx), 128, 0, stream>>>(contf, posfull, slen, atth);

        // out[b,q,h,:] = att @ V   (batched over b*H)
        gemm(atth, KVh + DHx, attoutf, Sx, DHx, Sx,
             Sx, 2*Dx, Dx,
             (long long)Hx*Sx*Sx, (long long)Sx*Sx,
             (long long)Sx*2*Dx, 2*DHx,
             (long long)Sx*Dx, DHx,
             Bx*Hx, Hx, 0);

        k_convert<<<grid1(BS*Dx), 256, 0, stream>>>(attoutf, attouth, BS*Dx);
        gemm(attouth, WoutT, aoutf, BS, Dx, Dx, Dx, Dx, Dx, 0,0,0,0,0,0, 1, 1, 1);

        // net = LN(X + attn)
        k_res_ln<<<dim3(BS), 256, 0, stream>>>(Xf, aoutf, ng, nb, netf, neth);

        // proj = tanh(relu(net@p1 + b1) @ p2 + b2)
        gemm(neth, p1T, ff1f, BS, DFFx, Dx, Dx, Dx, DFFx, 0,0,0,0,0,0, 1, 1, 1);
        k_bias_relu_bf16<<<grid1(BS*DFFx), 256, 0, stream>>>(ff1f, p1b, ff1h, BS*DFFx, DFFx);
        gemm(ff1h, p2T, t1f, BS, Dx, DFFx, DFFx, DFFx, Dx, 0,0,0,0,0,0, 1, 1, 1);
        k_bias_tanh<<<grid1(BS*Dx), 256, 0, stream>>>(t1f, p2b, projf, BS*Dx, Dx);

        // bgate = sigmoid(relu(net@g1 + b1) @ g2 + b2)
        gemm(neth, g1T, t1f, BS, Dx, Dx, Dx, Dx, Dx, 0,0,0,0,0,0, 1, 1, 1);
        k_bias_relu_bf16<<<grid1(BS*Dx), 256, 0, stream>>>(t1f, g1b, gateh, BS*Dx, Dx);
        gemm(gateh, g2T, t2f, BS, Dx, Dx, Dx, Dx, Dx, 0,0,0,0,0,0, 1, 1, 1);
        k_bias_sigm<<<grid1(BS*Dx), 256, 0, stream>>>(t2f, g2b, bgatef, BS*Dx, Dx);

        // src = src*(1-g) + proj*g
        k_gate<<<grid1(BS*Dx), 256, 0, stream>>>(Xf, projf, bgatef, BS*Dx);
    }

    k_copy<<<grid1(BS*Dx), 256, 0, stream>>>(Xf, (float*)d_out, BS*Dx);
}
